// LoRALinear_39934605918289
// MI455X (gfx1250) — matrix operand order repaired and checked by reading, not compiled, not run on hardware
//
#include <hip/hip_runtime.h>
#include <hip/hip_bf16.h>
#include <stdint.h>

typedef __attribute__((ext_vector_type(16))) __bf16       v16bf;
typedef __attribute__((ext_vector_type(8)))  float        v8f;
typedef __attribute__((ext_vector_type(4)))  float        v4f;
typedef __attribute__((ext_vector_type(4)))  unsigned int v4u;

union FragBF {
    v4u   u[2];
    v16bf v;
};

static __device__ __forceinline__ unsigned short f2bf(float f) {
    unsigned int u = __float_as_uint(f);
    u += 0x7FFFu + ((u >> 16) & 1u);   // round-to-nearest-even
    return (unsigned short)(u >> 16);
}

// Async DMA: global -> LDS, 16 bytes per lane, tracked by ASYNCcnt.
// LDS dest address = low 32 bits of the generic __shared__ pointer
// (shared-aperture encoding keeps the wave-relative LDS offset in addr[31:0]).
#define ASYNC_G2L_B128(gptr, lptr)                                          \
    asm volatile("global_load_async_to_lds_b128 %0, %1, off"                \
                 :: "v"((unsigned)(uintptr_t)(lptr)),                       \
                    "v"((unsigned long long)(uintptr_t)(gptr))              \
                 : "memory")

#define WAIT_ASYNCCNT_0()                                                   \
    asm volatile("s_wait_asynccnt 0x0" ::: "memory")

// ---------------------------------------------------------------------------
// Kernel 1: x (f32) -> bf16, 8 elems/thread. x read once -> NT loads.
// ---------------------------------------------------------------------------
__global__ __launch_bounds__(256)
void cvt_x_bf16(const float* __restrict__ x, unsigned short* __restrict__ xbf) {
    size_t i = ((size_t)blockIdx.x * 256u + threadIdx.x) * 8u;
    v4f f0 = __builtin_nontemporal_load((const v4f*)(x + i));
    v4f f1 = __builtin_nontemporal_load((const v4f*)(x + i + 4));
    v4u o;
    o.x = (unsigned)f2bf(f0.x) | ((unsigned)f2bf(f0.y) << 16);
    o.y = (unsigned)f2bf(f0.z) | ((unsigned)f2bf(f0.w) << 16);
    o.z = (unsigned)f2bf(f1.x) | ((unsigned)f2bf(f1.y) << 16);
    o.w = (unsigned)f2bf(f1.z) | ((unsigned)f2bf(f1.w) << 16);
    *(v4u*)(xbf + i) = o;                 // keep bf16 copy resident in L2
}

// ---------------------------------------------------------------------------
// Kernel 2: W_eff[n,k] = W[n,k] + 4 * sum_r B[n,r]*A[r,k]   -> bf16
// grid: (K/2048, N); W read once -> NT loads.
// ---------------------------------------------------------------------------
__global__ __launch_bounds__(256)
void make_weff_bf16(const float* __restrict__ W,
                    const float* __restrict__ A,   // (8, K)
                    const float* __restrict__ Bm,  // (N, 8)
                    unsigned short* __restrict__ wbf,
                    int K) {
    int n = blockIdx.y;
    int k = (blockIdx.x * 256 + threadIdx.x) * 8;
    size_t base = (size_t)n * K + k;
    v4f w0 = __builtin_nontemporal_load((const v4f*)(W + base));
    v4f w1 = __builtin_nontemporal_load((const v4f*)(W + base + 4));
    float acc[8] = {w0.x, w0.y, w0.z, w0.w, w1.x, w1.y, w1.z, w1.w};
    const float* brow = Bm + (size_t)n * 8;
#pragma unroll
    for (int r = 0; r < 8; ++r) {
        float b = 4.0f * brow[r];                     // scaling = alpha/r = 4
        const float* ar = A + (size_t)r * K + k;
        v4f a0 = *(const v4f*)(ar);
        v4f a1 = *(const v4f*)(ar + 4);
        acc[0] += b * a0.x; acc[1] += b * a0.y;
        acc[2] += b * a0.z; acc[3] += b * a0.w;
        acc[4] += b * a1.x; acc[5] += b * a1.y;
        acc[6] += b * a1.z; acc[7] += b * a1.w;
    }
    v4u o;
    o.x = (unsigned)f2bf(acc[0]) | ((unsigned)f2bf(acc[1]) << 16);
    o.y = (unsigned)f2bf(acc[2]) | ((unsigned)f2bf(acc[3]) << 16);
    o.z = (unsigned)f2bf(acc[4]) | ((unsigned)f2bf(acc[5]) << 16);
    o.w = (unsigned)f2bf(acc[6]) | ((unsigned)f2bf(acc[7]) << 16);
    *(v4u*)(wbf + base) = o;
}

// ---------------------------------------------------------------------------
// Kernel 3: out[m,n] = bias[n] + sum_k xbf[m,k] * wbf[n,k]
// Block tile 256x128, BK=32, 256 threads = 8 waves (4M x 2N),
// wave tile 64x64 = 4x4 WMMA tiles. Double-buffered LDS filled by
// GLOBAL_LOAD_ASYNC_TO_LDS_B128 (no VGPR staging), 1 barrier per K-step.
// ---------------------------------------------------------------------------
#define BM   256
#define BN   128
#define BK   32
#define LDSP 40                 // padded LDS row stride in bf16 elems (80 B)
#define ABUF (BM * LDSP)        // 10240 shorts per A buffer
#define BBUF (BN * LDSP)        // 5120 shorts per B buffer

__global__ __launch_bounds__(256)
void lora_gemm_bf16(const unsigned short* __restrict__ Abf,  // M x K (x)
                    const unsigned short* __restrict__ Bbf,  // N x K (W_eff)
                    const float* __restrict__ bias,
                    float* __restrict__ out,
                    int M, int N, int K) {
    __shared__ __align__(16) unsigned short As[2 * ABUF];
    __shared__ __align__(16) unsigned short Bs[2 * BBUF];

    const int tid  = threadIdx.x;
    const int lane = tid & 31;
    const int wid  = tid >> 5;
    const int wm   = wid & 3;        // wave M slot: 0..3 (64 rows each)
    const int wn   = wid >> 2;       // wave N slot: 0..1 (64 cols each)
    const int lm   = lane & 15;
    const int kh   = lane >> 4;      // half-wave selector

    const int m0 = blockIdx.y * BM;
    const int n0 = blockIdx.x * BN;

    // ---- per-thread staging descriptors (A: 1024 chunks, B: 512 chunks) ----
    const unsigned short* gA[4];
    int lA[4];
#pragma unroll
    for (int it = 0; it < 4; ++it) {
        int idx = tid + it * 256;
        int row = idx >> 2, c = idx & 3;
        gA[it] = Abf + (size_t)(m0 + row) * K + c * 8;
        lA[it] = row * LDSP + c * 8;
    }
    const unsigned short* gB[2];
    int lB[2];
#pragma unroll
    for (int it = 0; it < 2; ++it) {
        int idx = tid + it * 256;
        int row = idx >> 2, c = idx & 3;
        gB[it] = Bbf + (size_t)(n0 + row) * K + c * 8;
        lB[it] = row * LDSP + c * 8;
    }

    // ---- prologue: async-fill buffer 0 for kt = 0 ----
#pragma unroll
    for (int it = 0; it < 4; ++it) ASYNC_G2L_B128(gA[it], &As[lA[it]]);
#pragma unroll
    for (int it = 0; it < 2; ++it) ASYNC_G2L_B128(gB[it], &Bs[lB[it]]);
    WAIT_ASYNCCNT_0();
    __syncthreads();

    v8f zero;
#pragma unroll
    for (int e = 0; e < 8; ++e) zero[e] = 0.0f;
    v8f acc[4][4];
#pragma unroll
    for (int mt = 0; mt < 4; ++mt)
#pragma unroll
        for (int nt = 0; nt < 4; ++nt) acc[mt][nt] = zero;

    int p = 0;
    for (int kt = 0; kt < K; kt += BK) {
        const int ktn = kt + BK;
        const bool more = ktn < K;

        // ---- async DMA of next tile into buffer p^1; overlaps WMMAs ----
        if (more) {
            unsigned short* Aw = &As[(p ^ 1) * ABUF];
            unsigned short* Bw = &Bs[(p ^ 1) * BBUF];
#pragma unroll
            for (int it = 0; it < 4; ++it)
                ASYNC_G2L_B128(gA[it] + ktn, &Aw[lA[it]]);
#pragma unroll
            for (int it = 0; it < 2; ++it)
                ASYNC_G2L_B128(gB[it] + ktn, &Bw[lB[it]]);
        }

        // ---- fragments from buffer p (ISA 16-bit A 16x32 / B 32x16 layouts) ----
        const unsigned short* Ab = &As[p * ABUF];
        const unsigned short* Bb = &Bs[p * BBUF];
        FragBF af[4], bfg[4];
#pragma unroll
        for (int mt = 0; mt < 4; ++mt) {
            const unsigned short* q = Ab + (wm * 64 + mt * 16 + lm) * LDSP;
            af[mt].u[0] = *(const v4u*)(q + kh * 8);        // K = 8*kh .. +7
            af[mt].u[1] = *(const v4u*)(q + 16 + kh * 8);   // K = 16+8*kh .. +7
        }
#pragma unroll
        for (int nt = 0; nt < 4; ++nt) {
            const unsigned short* q = Bb + (wn * 64 + nt * 16 + lm) * LDSP;
            bfg[nt].u[0] = *(const v4u*)(q + kh * 8);       // K = 8*kh .. +7
            bfg[nt].u[1] = *(const v4u*)(q + 16 + kh * 8);  // K = 16+8*kh .. +7
        }

#pragma unroll
        for (int mt = 0; mt < 4; ++mt)
#pragma unroll
            for (int nt = 0; nt < 4; ++nt)
                acc[mt][nt] = __builtin_amdgcn_wmma_f32_16x16x32_bf16(
                    false, af[mt].v, false, bfg[nt].v,
                    (short)0, acc[mt][nt], false, false);

        // ---- drain this wave's DMA, then block-wide handoff ----
        if (more) WAIT_ASYNCCNT_0();
        __syncthreads();
        p ^= 1;
    }

    // ---- epilogue: C/D layout — lane: n = lm, row = v + 8*kh ----
    // out is write-once -> non-temporal stores keep the bf16 operands in L2.
#pragma unroll
    for (int nt = 0; nt < 4; ++nt) {
        int gn = n0 + wn * 64 + nt * 16 + lm;
        float bv = bias[gn];
#pragma unroll
        for (int mt = 0; mt < 4; ++mt) {
            int gm = m0 + wm * 64 + mt * 16 + kh * 8;
            float* op = out + (size_t)gm * N + gn;
#pragma unroll
            for (int v = 0; v < 8; ++v)
                __builtin_nontemporal_store(acc[mt][nt][v] + bv,
                                            op + (size_t)v * N);
        }
    }
}

// ---------------------------------------------------------------------------
extern "C" void kernel_launch(void* const* d_in, const int* in_sizes, int n_in,
                              void* d_out, int out_size, void* d_ws, size_t ws_size,
                              hipStream_t stream) {
    const float* x    = (const float*)d_in[0];   // (4, 2048, 4096)
    const float* W    = (const float*)d_in[1];   // (4096, 4096)
    const float* A    = (const float*)d_in[2];   // (8, 4096)
    const float* Bm   = (const float*)d_in[3];   // (4096, 8)
    const float* bias = (const float*)d_in[4];   // (4096,)
    float* out = (float*)d_out;

    const int K = in_sizes[2] / 8;               // 4096
    const int N = in_sizes[4];                   // 4096
    const int M = in_sizes[0] / K;               // 8192

    unsigned short* xbf = (unsigned short*)d_ws;          // M*K bf16 = 64 MiB
    unsigned short* wbf = xbf + (size_t)M * K;            // N*K bf16 = 32 MiB

    // 1) x -> bf16
    cvt_x_bf16<<<(unsigned)(((size_t)M * K) / (8 * 256)), 256, 0, stream>>>(x, xbf);

    // 2) W_eff = W + 4*(B@A) -> bf16   (folds LoRA into the weight once)
    dim3 gw(K / (8 * 256), N);
    make_weff_bf16<<<gw, 256, 0, stream>>>(W, A, Bm, wbf, K);

    // 3) main bf16 WMMA GEMM with bias epilogue
    dim3 gg(N / BN, M / BM);
    lora_gemm_bf16<<<gg, 256, 0, stream>>>(xbf, wbf, bias, out, M, N, K);
}
